// CfCTorch_3006477107700
// MI455X (gfx1250) — matrix operand order repaired and checked by reading, not compiled, not run on hardware
//
#include <hip/hip_runtime.h>
#include <math.h>

typedef __attribute__((ext_vector_type(16))) _Float16 v16h;
typedef __attribute__((ext_vector_type(8)))  _Float16 v8h;
typedef __attribute__((ext_vector_type(8)))  float    v8f;

#define B_ 256
#define S_ 2048
#define C_ 64
#define U_ 128
#define H_ 256
#define FEAT_LD 200   // 192 + 8 pad; row stride 400B (16B aligned)
#define FB_LD   264   // 256 + 8 pad; row stride 528B (16B aligned)
#define HS_LD   516   // 512 + 4 pad (f32)
#define TS_CH   128   // timespan chunk length (LDS-buffered)
#define TS_LD   132

#define LOG2E    1.4426950408889634f
#define TANH_K   1.9216698f          // 2 * 0.666 * log2(e)

// Load a 16x32 f16 A-tile (WMMA A layout) from row-major LDS.
// lane: row = lane&15, half h = lane>>4; elements 0..15 = K (32*kt + 16h + 0..15).
// Two ds_load_b128 per tile.
__device__ __forceinline__ v16h loadA(const _Float16* base, int ld, int kt,
                                      int row, int kofs) {
    const _Float16* p = base + row * ld + kt * 32 + kofs;
    v8h lo = *(const v8h*)(p);
    v8h hi = *(const v8h*)(p + 8);
    return __builtin_shufflevector(lo, hi, 0,1,2,3,4,5,6,7,8,9,10,11,12,13,14,15);
}

__device__ __forceinline__ v8f wmma_f16(v16h a, v16h b, v8f c) {
    return __builtin_amdgcn_wmma_f32_16x16x32_f16(false, a, false, b,
                                                  (short)0, c, false, false);
}

// branch-free lecun_tanh: 1.7159 * tanh(0.666x), tanh via exp2 + rcp (TRANS ops)
__device__ __forceinline__ float lecun_tanh_fast(float x) {
    float e = __builtin_amdgcn_exp2f(TANH_K * x);
    return 1.7159f * (1.0f - 2.0f * __builtin_amdgcn_rcpf(e + 1.0f));
}

__device__ __forceinline__ float sigmoid_neg_fast(float z) {
    // sigmoid(-z) = 1/(1+exp(z))
    return __builtin_amdgcn_rcpf(1.0f + __builtin_amdgcn_exp2f(LOG2E * z));
}

__global__ __launch_bounds__(512, 1)
void cfc_scan_kernel(const float* __restrict__ x_codes,
                     const float* __restrict__ h0,
                     const float* __restrict__ timespans,
                     const float* __restrict__ W0, const float* __restrict__ b0,
                     const float* __restrict__ W1, const float* __restrict__ b1,
                     const float* __restrict__ W2, const float* __restrict__ b2,
                     const float* __restrict__ Wa, const float* __restrict__ ba,
                     const float* __restrict__ Wb, const float* __restrict__ bb,
                     float* __restrict__ out)
{
    __shared__ _Float16 feat[16][FEAT_LD];   // [x_t | h], f16
    __shared__ _Float16 fbuf[16][FB_LD];     // backbone activation, f16
    __shared__ float    hstage[16][HS_LD];   // ff1|ff2|t_a|t_b, f32
    __shared__ float    tschunk[16][TS_LD];  // 128-step timespan chunk

    const int tid  = threadIdx.x;
    const int lane = tid & 31;
    const int wv   = tid >> 5;            // wave 0..15
    const int b0r  = blockIdx.x * 16;     // batch tile base row

    const int colq = lane & 15;           // col within 16-wide N tile / A row
    const int hh   = lane >> 4;           // lane half: 0 or 1
    const int kofs = hh << 4;             // A-layout K offset (0 / 16)
    const int k16  = hh << 4;             // B-layout K offset (0 / 16)

    // ------- one-time: load weights into registers in WMMA B layout -------
    v16h w0r[6];
    const int colB = wv * 16 + colq;
    #pragma unroll
    for (int k = 0; k < 6; ++k) {
        const int kb = k * 32 + k16;
        #pragma unroll
        for (int j = 0; j < 16; ++j)
            w0r[k][j] = (_Float16)W0[(size_t)(kb + j) * H_ + colB];
    }
    const float biasB = b0[colB];

    v16h  whr[2][8];
    float biasH[2];
    #pragma unroll
    for (int p = 0; p < 2; ++p) {
        const int g    = wv * 32 + p * 16 + colq;   // 0..511
        const int head = g >> 7;
        const int u    = g & 127;
        const float* W  = (head == 0) ? W1 : (head == 1) ? W2 : (head == 2) ? Wa : Wb;
        const float* bh = (head == 0) ? b1 : (head == 1) ? b2 : (head == 2) ? ba : bb;
        biasH[p] = bh[u];
        #pragma unroll
        for (int k = 0; k < 8; ++k) {
            const int kb = k * 32 + k16;
            #pragma unroll
            for (int j = 0; j < 16; ++j)
                whr[p][k][j] = (_Float16)W[(size_t)(kb + j) * U_ + u];
        }
    }

    // ------- init feat = [scaled x(t=0) | h0] -------
    for (int e = tid; e < 16 * C_; e += 512) {
        int r = e >> 6, c = e & 63;
        float xv = x_codes[(size_t)(b0r + r) * S_ * C_ + c];
        feat[r][c] = (_Float16)((xv - 65.0f) * 0.01f);
    }
    for (int e = tid; e < 16 * U_; e += 512) {
        int r = e >> 7, u = e & 127;
        feat[r][64 + u] = (_Float16)h0[(size_t)(b0r + r) * U_ + u];
    }
    __syncthreads();

    // ------- serial scan -------
    for (int t = 0; t < S_; ++t) {
        // refill 128-step timespan chunk (amortized: 16 refills total)
        if ((t & (TS_CH - 1)) == 0) {
            #pragma unroll
            for (int i = 0; i < 4; ++i) {
                int e = tid + 512 * i;            // 16*128 = 2048 elems
                int r = e >> 7, j = e & (TS_CH - 1);
                tschunk[r][j] = timespans[(size_t)(b0r + r) * S_ + t + j];
            }
            // phase-1 barrier below orders these writes before phase-3 reads
        }

        // prefetch next x (raw floats; consumed only in phase 3 so the
        // s_wait_loadcnt sinks past both GEMM phases)
        float xp0 = 0.f, xp1 = 0.f;
        if (t + 1 < S_) {
            int e = tid * 2;
            int r = e >> 6, c = e & 63;
            const float* xp = x_codes + (size_t)(b0r + r) * S_ * C_ +
                              (size_t)(t + 1) * C_ + c;
            xp0 = xp[0];
            xp1 = xp[1];
        }

        // ---- phase 1: backbone GEMM [16,192]x[192,256] + lecun_tanh ----
        v8f acc = {};
        {
            v16h acur = loadA(&feat[0][0], FEAT_LD, 0, colq, kofs);
            #pragma unroll
            for (int k = 0; k < 6; ++k) {
                v16h anext = acur;
                if (k < 5) anext = loadA(&feat[0][0], FEAT_LD, k + 1, colq, kofs);
                acc = wmma_f16(acur, w0r[k], acc);
                acur = anext;
            }
        }
        #pragma unroll
        for (int v = 0; v < 8; ++v) {
            float y = lecun_tanh_fast(acc[v] + biasB);
            fbuf[v + (hh << 3)][colB] = (_Float16)y;   // D: row = v+8*hh
        }
        __syncthreads();

        // ---- phase 2: 4 heads fused as one [16,256]x[256,512] GEMM ----
        v8f hacc0 = {}, hacc1 = {};
        {
            v16h acur = loadA(&fbuf[0][0], FB_LD, 0, colq, kofs);
            #pragma unroll
            for (int k = 0; k < 8; ++k) {
                v16h anext = acur;
                if (k < 7) anext = loadA(&fbuf[0][0], FB_LD, k + 1, colq, kofs);
                hacc0 = wmma_f16(acur, whr[0][k], hacc0);
                hacc1 = wmma_f16(acur, whr[1][k], hacc1);
                acur = anext;
            }
        }
        {
            const int g0 = wv * 32 + colq;
            #pragma unroll
            for (int v = 0; v < 8; ++v) {
                hstage[v + (hh << 3)][g0]      = hacc0[v] + biasH[0];
                hstage[v + (hh << 3)][g0 + 16] = hacc1[v] + biasH[1];
            }
        }
        __syncthreads();

        // ---- phase 3: gate + blend, write out, update h in feat ----
        #pragma unroll
        for (int i = 0; i < 4; ++i) {
            int e = tid + 512 * i;           // 0..2047
            int r = e >> 7, u = e & 127;
            float f1 = hstage[r][u];
            float f2 = hstage[r][128 + u];
            float ta = hstage[r][256 + u];
            float tb = hstage[r][384 + u];
            float s  = sigmoid_neg_fast(ta * tschunk[r][t & (TS_CH - 1)] - tb);
            float nh = f1 + s * (f2 - f1);
            out[((size_t)(b0r + r) * S_ + t) * U_ + u] = nh;
            feat[r][64 + u] = (_Float16)nh;
        }
        if (t + 1 < S_) {
            // pin raw x values here: prevents the scheduler from hoisting the
            // conversion (and its s_wait_loadcnt) above the GEMM phases
            asm volatile("" : "+v"(xp0), "+v"(xp1));
            int e = tid * 2;
            int r = e >> 6, c = e & 63;
            feat[r][c]     = (_Float16)((xp0 - 65.0f) * 0.01f);
            feat[r][c + 1] = (_Float16)((xp1 - 65.0f) * 0.01f);
        }
        __syncthreads();
    }
}

extern "C" void kernel_launch(void* const* d_in, const int* in_sizes, int n_in,
                              void* d_out, int out_size, void* d_ws, size_t ws_size,
                              hipStream_t stream) {
    (void)in_sizes; (void)n_in; (void)out_size; (void)d_ws; (void)ws_size;
    const float* x  = (const float*)d_in[0];
    const float* h0 = (const float*)d_in[1];
    const float* ts = (const float*)d_in[2];
    const float* W0 = (const float*)d_in[3];
    const float* b0 = (const float*)d_in[4];
    const float* W1 = (const float*)d_in[5];
    const float* b1 = (const float*)d_in[6];
    const float* W2 = (const float*)d_in[7];
    const float* b2 = (const float*)d_in[8];
    const float* Wa = (const float*)d_in[9];
    const float* ba = (const float*)d_in[10];
    const float* Wb = (const float*)d_in[11];
    const float* bb = (const float*)d_in[12];
    float* out = (float*)d_out;

    dim3 grid(B_ / 16);   // 16 persistent workgroups, one batch tile each
    dim3 block(512);      // 16 wave32s
    hipLaunchKernelGGL(cfc_scan_kernel, grid, block, 0, stream,
                       x, h0, ts, W0, b0, W1, b1, W2, b2, Wa, ba, Wb, bb, out);
}
